// Attention1D_77524159693259
// MI455X (gfx1250) — matrix operand order repaired and checked by reading, not compiled, not run on hardware
//
#include <hip/hip_runtime.h>
#include <stdint.h>
#include <stddef.h>

// Problem dims (match reference)
#define BS   8
#define TT   1024
#define DIN  64
#define NHH  8
#define DOUTT 1024
#define RDIM 512   // NH*DIN

typedef __bf16 bf16_t;
typedef __attribute__((ext_vector_type(16))) __bf16 v16bf;
typedef __attribute__((ext_vector_type(8)))  __bf16 v8bf;
typedef __attribute__((ext_vector_type(8)))  float  v8f;

// ---------------------------------------------------------------------------
// Kernel 0: q/k projections.
//   q[b,t,j]  = X[b,t,:]·Wq_w[j,:] + Wq_b[j]        (row-major, j fast)
//   kT[b,j,l] = X[b,l,:]·Wk_w[j,:] + Wk_b[j]        (TRANSPOSED: l fast, so
//                                                    the scores wave reads k
//                                                    fully coalesced)
// ---------------------------------------------------------------------------
__global__ void qk_kernel(const float* __restrict__ X,
                          const float* __restrict__ Wq_w, const float* __restrict__ Wq_b,
                          const float* __restrict__ Wk_w, const float* __restrict__ Wk_b,
                          float* __restrict__ q, float* __restrict__ kT) {
  int idx = blockIdx.x * blockDim.x + threadIdx.x;
  if (idx >= BS * TT * 16) return;
  int j  = idx & 15;
  int bt = idx >> 4;
  int t  = bt & (TT - 1);
  int b  = bt >> 10;
  const float* x  = X + (size_t)bt * DIN;
  const float* wq = Wq_w + j * DIN;
  const float* wk = Wk_w + j * DIN;
  float aq = Wq_b[j], ak = Wk_b[j];
#pragma unroll
  for (int d = 0; d < DIN; ++d) {
    float xv = x[d];
    aq += xv * wq[d];
    ak += xv * wk[d];
  }
  q[idx] = aq;
  kT[((size_t)(b * 16 + j)) * TT + t] = ak;
}

// ---------------------------------------------------------------------------
// Kernel 0b: XT[b][d][l] = bf16(X[b][l][d])   (B-major layout for gemm1)
// ---------------------------------------------------------------------------
__global__ void xt_kernel(const float* __restrict__ X, bf16_t* __restrict__ XT) {
  int idx = blockIdx.x * blockDim.x + threadIdx.x;
  if (idx >= BS * DIN * TT) return;
  int l = idx & (TT - 1);
  int d = (idx >> 10) & (DIN - 1);
  int b = idx >> 16;
  XT[idx] = (bf16_t)X[((size_t)(b * TT + l)) * DIN + d];
}

// ---------------------------------------------------------------------------
// Kernel 0c: Wbf = bf16(W_w)  (W_w is (d_out, T) = already B-major for gemm2)
// ---------------------------------------------------------------------------
__global__ void wbf_kernel(const float* __restrict__ W, bf16_t* __restrict__ Wbf) {
  int idx = blockIdx.x * blockDim.x + threadIdx.x;
  if (idx >= DOUTT * TT) return;
  Wbf[idx] = (bf16_t)W[idx];
}

// ---------------------------------------------------------------------------
// Kernel 1: fused scores + softmax -> P_bf16[(b*8+h)*T + t][l]
// One wave32 per (b,h,t) row; each lane owns 32 of the 1024 l positions.
// s = (q+u[t])·k[b,l,h] + cA*δ² + cB*δ   (δ = l - t)
// ---------------------------------------------------------------------------
__global__ void scores_kernel(const float* __restrict__ q, const float* __restrict__ kT,
                              const float* __restrict__ Wkey, const float* __restrict__ u,
                              const float* __restrict__ ac, const float* __restrict__ alpha,
                              bf16_t* __restrict__ P) {
  int wave = threadIdx.x >> 5;
  int lane = threadIdx.x & 31;
  int row  = blockIdx.x * 8 + wave;        // (b*8 + h)*1024 + t
  int t    = row & (TT - 1);
  int bh   = row >> 10;
  int h    = bh & 7;
  int b    = bh >> 3;

  float a   = alpha[0];
  size_t qi = ((size_t)(b * TT + t)) * 16 + h * 2;
  float q0  = q[qi + 0], q1 = q[qi + 1];
  float qe0 = q0 + u[t * 2 + 0];
  float qe1 = q1 + u[t * 2 + 1];
  float qv0 = q0 - a;                      // v[h] = (-alpha, 2*alpha*ac[h])
  float qv1 = q1 + 2.0f * a * ac[h];
  float cA  = qv0 * Wkey[0] + qv1 * Wkey[2];   // coeff of δ²
  float cB  = qv0 * Wkey[1] + qv1 * Wkey[3];   // coeff of δ

  const float* kb0 = kT + ((size_t)(b * 16 + h * 2 + 0)) * TT;  // coalesced
  const float* kb1 = kT + ((size_t)(b * 16 + h * 2 + 1)) * TT;

  float s[32];
  float m = -3.402823466e38f;
#pragma unroll
  for (int i = 0; i < 32; ++i) {
    int l = i * 32 + lane;
    float k0 = kb0[l];
    float k1 = kb1[l];
    float dl = (float)(l - t);
    float sv = qe0 * k0 + qe1 * k1 + (cA * dl + cB) * dl;
    s[i] = sv;
    m = fmaxf(m, sv);
  }
#pragma unroll
  for (int off = 16; off > 0; off >>= 1) m = fmaxf(m, __shfl_xor(m, off, 32));

  float sum = 0.0f;
#pragma unroll
  for (int i = 0; i < 32; ++i) { s[i] = __expf(s[i] - m); sum += s[i]; }
#pragma unroll
  for (int off = 16; off > 0; off >>= 1) sum += __shfl_xor(sum, off, 32);
  float inv = 1.0f / sum;

  bf16_t* prow = P + (size_t)row * TT;
#pragma unroll
  for (int i = 0; i < 32; ++i) prow[i * 32 + lane] = (bf16_t)(s[i] * inv);
}

// ---------------------------------------------------------------------------
// WMMA fragment loaders (ISA §7.12.2 16-bit A 16x32 / B 32x16 layouts, wave32)
//   A: lane holds row M=lane&15; K pairs: j<4 -> K=half*8+2j(+1), j>=4 -> +16
//      => two contiguous 8-element (16B) segments at k0+half*8 and k0+16+half*8
//   B: lane holds col N=lane&15; same K order as A (segments at k0+half*8, k0+16+half*8)
// ---------------------------------------------------------------------------
__device__ __forceinline__ v16bf load_a_frag(const bf16_t* row, int k0, int half8) {
  v16bf a;
  *(v8bf*)&a       = *(const v8bf*)(row + k0 + half8);
  *((v8bf*)&a + 1) = *(const v8bf*)(row + k0 + 16 + half8);
  return a;
}
__device__ __forceinline__ v16bf load_b_frag(const bf16_t* col, int k0, int half16) {
  v16bf bb;
  *(v8bf*)&bb       = *(const v8bf*)(col + k0 + half16);
  *((v8bf*)&bb + 1) = *(const v8bf*)(col + k0 + 16 + half16);
  return bb;
}

// ---------------------------------------------------------------------------
// Kernel 2: gemm1  iv[t,d] = sum_l P[bh][t][l] * X[b][l][d]   (per b,h)
// One wave computes a 16(t) x 64(d) tile = 4 accumulators; stores D transposed
// as bf16 into ivT[b][h*64+d][t] (A-major layout for gemm2).
// P is the only HBM-streamed operand (128 MB) -> prefetch it ahead.
// ---------------------------------------------------------------------------
__global__ void gemm1_kernel(const bf16_t* __restrict__ P, const bf16_t* __restrict__ XT,
                             bf16_t* __restrict__ ivT) {
  int wave  = (blockIdx.x * blockDim.x + threadIdx.x) >> 5;
  int lane  = threadIdx.x & 31;
  int ttile = wave & 63;           // 64 t-tiles of 16
  int bh    = wave >> 6;           // 0..63
  int b     = bh >> 3;
  int h     = bh & 7;
  int t0    = ttile * 16;
  int laneLo = lane & 15;
  int half   = lane >> 4;

  const bf16_t* arow  = P + ((size_t)bh * TT + (t0 + laneLo)) * TT;
  const bf16_t* bbase = XT + (size_t)b * DIN * TT;          // [d][l]

  v8f acc[4] = {};
  for (int k0 = 0; k0 < TT; k0 += 32) {
    // prefetch the A stream one unrolled iteration (128 elems = 256 B) ahead
    __builtin_prefetch(arow + k0 + 128, 0, 0);
    v16bf a = load_a_frag(arow, k0, half * 8);
#pragma unroll
    for (int nt = 0; nt < 4; ++nt) {
      const bf16_t* bcol = bbase + (size_t)(nt * 16 + laneLo) * TT;
      v16bf bb = load_b_frag(bcol, k0, half * 8);
      acc[nt] = __builtin_amdgcn_wmma_f32_16x16x32_bf16(
          false, a, false, bb, (short)0, acc[nt], false, false);
    }
  }
  // D layout: VGPR i -> (M = i + 8*half, N = lane&15). Store transposed.
#pragma unroll
  for (int nt = 0; nt < 4; ++nt) {
    int r = h * 64 + nt * 16 + laneLo;
#pragma unroll
    for (int i = 0; i < 8; ++i) {
      int tt = t0 + i + half * 8;
      ivT[((size_t)(b * RDIM + r)) * TT + tt] = (bf16_t)acc[nt][i];
    }
  }
}

// ---------------------------------------------------------------------------
// Kernel 3: gemm2  out[b][r][o] = sum_t ivT[b][r][t] * W_w[o][t] + W_b[o]
// One wave computes a 16(r) x 64(o) tile; f32 output.
// ---------------------------------------------------------------------------
__global__ void gemm2_kernel(const bf16_t* __restrict__ ivT, const bf16_t* __restrict__ Wbf,
                             const float* __restrict__ W_b, float* __restrict__ out) {
  int wave = (blockIdx.x * blockDim.x + threadIdx.x) >> 5;
  int lane = threadIdx.x & 31;
  int b    = wave >> 9;            // 512 waves per batch
  int rem  = wave & 511;
  int mt   = rem >> 4;             // 32 r-tiles of 16
  int ng   = rem & 15;             // 16 o-groups of 64
  int r0   = mt * 16;
  int o0   = ng * 64;
  int laneLo = lane & 15;
  int half   = lane >> 4;

  const bf16_t* arow = ivT + ((size_t)(b * RDIM + r0 + laneLo)) * TT;

  v8f acc[4] = {};
  for (int k0 = 0; k0 < TT; k0 += 32) {
    __builtin_prefetch(arow + k0 + 128, 0, 0);
    v16bf a = load_a_frag(arow, k0, half * 8);
#pragma unroll
    for (int nt = 0; nt < 4; ++nt) {
      const bf16_t* bcol = Wbf + (size_t)(o0 + nt * 16 + laneLo) * TT;
      v16bf bb = load_b_frag(bcol, k0, half * 8);
      acc[nt] = __builtin_amdgcn_wmma_f32_16x16x32_bf16(
          false, a, false, bb, (short)0, acc[nt], false, false);
    }
  }
#pragma unroll
  for (int nt = 0; nt < 4; ++nt) {
    int o    = o0 + nt * 16 + laneLo;
    float bw = W_b[o];
#pragma unroll
    for (int i = 0; i < 8; ++i) {
      int r = r0 + i + half * 8;
      out[((size_t)(b * RDIM + r)) * DOUTT + o] = acc[nt][i] + bw;
    }
  }
}

// ---------------------------------------------------------------------------
// Launch.  Workspace layout (bytes), needs ~147 MB:
//   P    @ 0          : 8*8*1024*1024*2 = 134217728
//   q    @ 134217728  : 8*1024*16*4     =    524288
//   kT   @ 134742016  : 8*16*1024*4     =    524288
//   XT   @ 135266304  : 8*64*1024*2     =   1048576
//   Wbf  @ 136314880  : 1024*1024*2     =   2097152
//   ivT  @ 138412032  : 8*512*1024*2    =   8388608
// ---------------------------------------------------------------------------
extern "C" void kernel_launch(void* const* d_in, const int* in_sizes, int n_in,
                              void* d_out, int out_size, void* d_ws, size_t ws_size,
                              hipStream_t stream) {
  const float* X     = (const float*)d_in[0];
  const float* Wq_w  = (const float*)d_in[1];
  const float* Wq_b  = (const float*)d_in[2];
  const float* Wk_w  = (const float*)d_in[3];
  const float* Wk_b  = (const float*)d_in[4];
  const float* Wkey  = (const float*)d_in[5];
  const float* u     = (const float*)d_in[6];
  // d_in[7] = R (recomputed analytically from δ = l - t)
  const float* ac    = (const float*)d_in[8];
  const float* alpha = (const float*)d_in[9];
  const float* W_w   = (const float*)d_in[10];
  const float* W_b   = (const float*)d_in[11];
  float* out = (float*)d_out;

  char* ws = (char*)d_ws;
  bf16_t* P   = (bf16_t*)(ws);
  float*  q   = (float*)(ws + 134217728);
  float*  kT  = (float*)(ws + 134742016);
  bf16_t* XT  = (bf16_t*)(ws + 135266304);
  bf16_t* Wbf = (bf16_t*)(ws + 136314880);
  bf16_t* ivT = (bf16_t*)(ws + 138412032);

  // Projections + bf16 staging
  qk_kernel<<<(BS * TT * 16 + 255) / 256, 256, 0, stream>>>(X, Wq_w, Wq_b, Wk_w, Wk_b, q, kT);
  xt_kernel<<<(BS * DIN * TT + 255) / 256, 256, 0, stream>>>(X, XT);
  wbf_kernel<<<(DOUTT * TT + 255) / 256, 256, 0, stream>>>(W_w, Wbf);

  // Fused scores + softmax (one wave per row; 65536 rows, 8 rows/block)
  scores_kernel<<<(BS * NHH * TT) / 8, 256, 0, stream>>>(q, kT, Wkey, u, ac, alpha, P);

  // GEMM1: 64 (b,h) x 64 t-tiles = 4096 waves -> 512 blocks of 256
  gemm1_kernel<<<512, 256, 0, stream>>>(P, XT, ivT);

  // GEMM2: 8 b x 512 tiles = 4096 waves -> 512 blocks of 256
  gemm2_kernel<<<512, 256, 0, stream>>>(ivT, Wbf, W_b, out);
}
